// SocialAttention_8340826489419
// MI455X (gfx1250) — matrix operand order repaired and checked by reading, not compiled, not run on hardware
//
#include <hip/hip_runtime.h>
#include <hip/hip_bf16.h>

// ---------------------------------------------------------------------------
// SocialAttention fused bf16-WMMA implementation for gfx1250 (MI455X).
//   Layer1: x[64x96(pad)] @ W1T -> h[64x512]   (v_wmma_f32_16x16x32_bf16)
//   Layer2: h[64x512]     @ W2T -> h2[64x32]   (v_wmma_f32_16x16x32_bf16)
//   Attention/softmax/pool in VALU, 4 scenes per workgroup to amortize W_attn.
// ---------------------------------------------------------------------------

#define P_   64
#define H_   64
#define E_   16
#define B_   32
#define HID_ 512
#define KX_  96            // layer-1 K padded 80 -> 96 (3 k-steps of 32)
#define SPW_ 4             // scenes per workgroup

typedef __bf16 v16bf  __attribute__((ext_vector_type(16)));
typedef __bf16 bf16x8 __attribute__((ext_vector_type(8)));
typedef __bf16 bf16x4 __attribute__((ext_vector_type(4)));
typedef float  v8f    __attribute__((ext_vector_type(8)));

union ABu { bf16x8 h[2]; v16bf v; };

// ---------------------------------------------------------------------------
// Prologue: transpose + bf16-convert weights into workspace.
//   w1t[n*96 + k] = bf16(W1[k*512 + n]) (k<80) else 0     (512x96)
//   w2t[n*512 + k] = bf16(W2[k*32 + n])                   (32x512)
// ---------------------------------------------------------------------------
__global__ __launch_bounds__(256) void social_convert_weights(
    const float* __restrict__ W1, const float* __restrict__ W2,
    __bf16* __restrict__ w1t, __bf16* __restrict__ w2t) {
  int i = blockIdx.x * 256 + threadIdx.x;
  if (i < HID_ * KX_) {                         // 49152 elements
    int n = i / KX_, k = i - n * KX_;
    w1t[i] = (k < (H_ + E_)) ? (__bf16)W1[k * HID_ + n] : (__bf16)0.0f;
  } else {
    int j = i - HID_ * KX_;                     // 16384 elements
    if (j < B_ * HID_) {
      int n = j / HID_, k = j - n * HID_;
      w2t[j] = (__bf16)W2[k * B_ + n];
    }
  }
}

// ---------------------------------------------------------------------------
// Main fused kernel: 1024 workgroups x 256 threads (8 wave32), 4 scenes each.
// ---------------------------------------------------------------------------
__global__ __launch_bounds__(256) void social_main(
    const float* __restrict__ h_states, const float* __restrict__ end_pos,
    const float* __restrict__ W_spat,  const float* __restrict__ b_spat,
    const float* __restrict__ b1,      const float* __restrict__ b2,
    const float* __restrict__ W_attn,  const float* __restrict__ b_attn,
    const __bf16* __restrict__ w1t,
    const __bf16* __restrict__ w2t,
    float* __restrict__ out) {
  extern __shared__ char smem[];
  __bf16* xbf = (__bf16*)smem;                          // 64*96  bf16 = 12288 B
  __bf16* hbf = (__bf16*)(smem + 12288);                // 64*512 bf16 = 65536 B
  float* h2       = (float*)(smem + 12288 + 65536);     // 4*64*32 f32 = 32768 B
  float* partials = h2 + SPW_ * P_ * B_;                // 4*256 f32   =  4096 B
  float* logits   = partials + SPW_ * 256;              // 4*64 f32
  float* wexp     = logits + SPW_ * P_;                 // 4*64 f32

  const int tid  = threadIdx.x;
  const int wave = tid >> 5;
  const int lane = tid & 31;
  const int half = lane >> 4;   // 0: lanes 0-15, 1: lanes 16-31
  const int lr   = lane & 15;

  for (int s = 0; s < SPW_; ++s) {
    const int n = blockIdx.x * SPW_ + s;

    // ---- Stage A: build x = [h_states | spatial_embed | 0-pad] in bf16 LDS
    const float4* hs4 = (const float4*)(h_states + (size_t)n * P_ * H_);
    for (int i = tid; i < (P_ * H_) / 4; i += 256) {   // 1024 float4s
      float4 v = hs4[i];
      int p = i >> 4, c = (i & 15) << 2;
      bf16x4 o;
      o[0] = (__bf16)v.x; o[1] = (__bf16)v.y;
      o[2] = (__bf16)v.z; o[3] = (__bf16)v.w;
      *(bf16x4*)(xbf + p * KX_ + c) = o;
    }
    const float* ep = end_pos + (size_t)n * P_ * 2;
    const float ex0 = ep[0], ey0 = ep[1];
    for (int i = tid; i < P_ * 32; i += 256) {
      int p = i >> 5, c = i & 31;
      __bf16 v = (__bf16)0.0f;
      if (c < E_) {
        float rx = ep[p * 2]     - ex0;
        float ry = ep[p * 2 + 1] - ey0;
        v = (__bf16)(b_spat[c] + rx * W_spat[c] + ry * W_spat[E_ + c]);
      }
      xbf[p * KX_ + H_ + c] = v;
    }
    __syncthreads();

    // ---- Layer 1: x[64x96] @ W1T -> relu -> hbf[64x512] (bf16)
    // Wave w owns N-tiles {w, w+8, w+16, w+24}; B fragments loaded once per
    // N-tile and reused across all 4 M-tiles (4x fewer global loads).
    for (int q = 0; q < 4; ++q) {
      const int nt = wave + q * 8;
      const int ncol = nt * 16 + lr;
      const __bf16* brow = w1t + ncol * KX_;
      const v16bf bf0 = *(const v16bf*)(brow +  0 + half * 16);
      const v16bf bf1 = *(const v16bf*)(brow + 32 + half * 16);
      const v16bf bf2 = *(const v16bf*)(brow + 64 + half * 16);
      const float bias = b1[ncol];
#pragma unroll
      for (int mt = 0; mt < 4; ++mt) {
        const __bf16* arow = xbf + (mt * 16 + lr) * KX_;
        ABu a0, a1, a2;
        a0.h[0] = *(const bf16x8*)(arow +  0 + half * 16);
        a0.h[1] = *(const bf16x8*)(arow +  8 + half * 16);
        a1.h[0] = *(const bf16x8*)(arow + 32 + half * 16);
        a1.h[1] = *(const bf16x8*)(arow + 40 + half * 16);
        a2.h[0] = *(const bf16x8*)(arow + 64 + half * 16);
        a2.h[1] = *(const bf16x8*)(arow + 72 + half * 16);
        v8f acc = {};
        acc = __builtin_amdgcn_wmma_f32_16x16x32_bf16(
            false, a0.v, false, bf0, (short)0, acc, false, false);
        acc = __builtin_amdgcn_wmma_f32_16x16x32_bf16(
            false, a1.v, false, bf1, (short)0, acc, false, false);
        acc = __builtin_amdgcn_wmma_f32_16x16x32_bf16(
            false, a2.v, false, bf2, (short)0, acc, false, false);
#pragma unroll
        for (int r = 0; r < 8; ++r) {
          float v = fmaxf(acc[r] + bias, 0.0f);
          hbf[(mt * 16 + r + half * 8) * HID_ + ncol] = (__bf16)v;
        }
      }
    }
    __syncthreads();

    // ---- Layer 2: h[64x512] @ W2T -> relu -> h2[s][64x32] (f32 LDS)
    // 4 M-tiles x 2 N-tiles = 8 tiles = 1 per wave, 16 k-steps.
    {
      const int mt = wave >> 1, nt = wave & 1;
      const int ncol = nt * 16 + lr;
      v8f acc = {};
      const __bf16* arow = hbf + (mt * 16 + lr) * HID_;
      const __bf16* brow = w2t + ncol * HID_;
#pragma unroll 4
      for (int kt = 0; kt < 16; ++kt) {
        const int k0 = kt * 32;
        ABu a;
        a.h[0] = *(const bf16x8*)(arow + k0 + half * 16);
        a.h[1] = *(const bf16x8*)(arow + k0 + 8 + half * 16);
        v16bf bfrag = *(const v16bf*)(brow + k0 + half * 16);
        acc = __builtin_amdgcn_wmma_f32_16x16x32_bf16(
            false, a.v, false, bfrag, (short)0, acc, false, false);
      }
      const float bias = b2[ncol];
#pragma unroll
      for (int r = 0; r < 8; ++r) {
        h2[s * (P_ * B_) + (mt * 16 + r + half * 8) * B_ + ncol] =
            fmaxf(acc[r] + bias, 0.0f);
      }
    }
    __syncthreads();
  }

  // ---- Attention logits for all 4 scenes in one sweep over W_attn.
  // thread t = (seg, pp): seg = t>>6 covers i in [seg*512, seg*512+512).
  // Within a wave, h2[s][i] is an LDS broadcast; W_attn is lane-coalesced.
  {
    const int pp = tid & 63, seg = tid >> 6;
    float p0 = 0.f, p1 = 0.f, p2 = 0.f, p3 = 0.f;
    const int i0 = seg * 512;
    for (int i = i0; i < i0 + 512; ++i) {
      float w = W_attn[(size_t)i * P_ + pp];   // reused by 4 scenes
      p0 += h2[0 * 2048 + i] * w;
      p1 += h2[1 * 2048 + i] * w;
      p2 += h2[2 * 2048 + i] * w;
      p3 += h2[3 * 2048 + i] * w;
    }
    partials[0 * 256 + tid] = p0;
    partials[1 * 256 + tid] = p1;
    partials[2 * 256 + tid] = p2;
    partials[3 * 256 + tid] = p3;
  }
  __syncthreads();

  // logits[s][p] = b_attn[p] + sum over 4 segments
  {
    int s = tid >> 6, p = tid & 63;
    float l = b_attn[p];
#pragma unroll
    for (int g = 0; g < 4; ++g) l += partials[s * 256 + g * 64 + p];
    logits[s * 64 + p] = l;
  }
  __syncthreads();

  // softmax numerator (max-subtracted exp), one (s,p) per thread
  {
    int s = tid >> 6;
    float mx = -3.402823466e38f;
    for (int q = 0; q < 64; ++q) mx = fmaxf(mx, logits[s * 64 + q]);
    wexp[tid] = __expf(logits[tid] - mx);
  }
  __syncthreads();

  // pooling: out[n][b] = sum_p h2[p][b] * softmax_p ; 4 scenes x 32 outputs
  if (tid < SPW_ * B_) {
    int s = tid >> 5, b = tid & 31;
    float ssum = 0.f;
    for (int q = 0; q < 64; ++q) ssum += wexp[s * 64 + q];
    float acc = 0.f;
    for (int q = 0; q < 64; ++q)
      acc += h2[s * 2048 + q * B_ + b] * wexp[s * 64 + q];
    int n = blockIdx.x * SPW_ + s;
    out[(size_t)n * B_ + b] = acc / ssum;
  }
}

// ---------------------------------------------------------------------------
// Launch
// ---------------------------------------------------------------------------
extern "C" void kernel_launch(void* const* d_in, const int* in_sizes, int n_in,
                              void* d_out, int out_size, void* d_ws, size_t ws_size,
                              hipStream_t stream) {
  const float* h_states = (const float*)d_in[0];
  const float* end_pos  = (const float*)d_in[1];
  const float* W_spat   = (const float*)d_in[2];
  const float* b_spat   = (const float*)d_in[3];
  const float* W1       = (const float*)d_in[4];
  const float* b1       = (const float*)d_in[5];
  const float* W2       = (const float*)d_in[6];
  const float* b2       = (const float*)d_in[7];
  const float* W_attn   = (const float*)d_in[8];
  const float* b_attn   = (const float*)d_in[9];
  float* out = (float*)d_out;

  // workspace: W1T bf16 (512*96) then W2T bf16 (32*512); 128 KiB total
  __bf16* w1t = (__bf16*)d_ws;
  __bf16* w2t = w1t + HID_ * KX_;

  social_convert_weights<<<(HID_ * KX_ + B_ * HID_ + 255) / 256, 256, 0, stream>>>(
      W1, W2, w1t, w2t);

  const size_t smem = 12288 /*xbf*/ + 65536 /*hbf*/ + 32768 /*h2*/ +
                      4096 /*partials*/ + 1024 /*logits*/ + 1024 /*wexp*/;
  social_main<<<4096 / SPW_, 256, smem, stream>>>(
      h_states, end_pos, W_spat, b_spat, b1, b2, W_attn, b_attn, w1t, w2t, out);
}
